// MultiHeadAttention_14834817040607
// MI455X (gfx1250) — matrix operand order repaired and checked by reading, not compiled, not run on hardware
//
#include <hip/hip_runtime.h>
#include <hip/hip_bf16.h>

// MI455X (gfx1250) MHA: bf16 WMMA GEMMs (64x32/wave, no spills) + flash attention
// with double-buffered async-to-LDS K/V staging. B=2, S=2048, D_MODEL=1024, H=16, DK=64.

#define B_  2
#define S_  2048
#define DM  1024
#define H_  16
#define DK  64
#define NT  (B_ * S_)          // 4096 tokens

typedef __attribute__((ext_vector_type(16))) __bf16 v16bf;
typedef __attribute__((ext_vector_type(8)))  float  v8f;

struct alignas(16) B16x8 { __bf16 h[8]; };

#define WMMA_BF16(a, b, c) \
  __builtin_amdgcn_wmma_f32_16x16x32_bf16(false, (a), false, (b), (short)0, (c), false, false)

// ---- fragment loaders (contiguous k = 16*half + e, same order for A and B) ----

// A 16x32 bf16: lane = row M; elements e = K kBase+e (16 contiguous),
// kBase = (lane<16)?0:16.  Caller pre-offsets p by row*ld + kBase.
__device__ inline v16bf ld_fragA(const __bf16* p) {
  B16x8 lo = *(const B16x8*)(p);
  B16x8 hi = *(const B16x8*)(p + 8);
  v16bf a;
#pragma unroll
  for (int i = 0; i < 8; i++) { a[i] = lo.h[i]; a[i + 8] = hi.h[i]; }
  return a;
}

// B 32x16 bf16: lane = column N; elements e = K half*16+e (16 contiguous).
__device__ inline v16bf ld_fragB(const __bf16* p) {
  B16x8 lo = *(const B16x8*)(p);
  B16x8 hi = *(const B16x8*)(p + 8);
  v16bf b;
#pragma unroll
  for (int i = 0; i < 8; i++) { b[i] = lo.h[i]; b[i + 8] = hi.h[i]; }
  return b;
}

__device__ inline float rmax16(float v) {
#pragma unroll
  for (int m = 1; m < 16; m <<= 1) v = fmaxf(v, __shfl_xor(v, m, 32));
  return v;
}
__device__ inline float rsum16(float v) {
#pragma unroll
  for (int m = 1; m < 16; m <<= 1) v += __shfl_xor(v, m, 32);
  return v;
}

// ---- gfx1250 async global->LDS copy (ASYNCcnt tracked) ----
__device__ inline void async_copy_b128(unsigned lds_byte_addr, const void* gsrc) {
  asm volatile("global_load_async_to_lds_b128 %0, %1, off"
               :: "v"(lds_byte_addr), "v"((unsigned long long)(uintptr_t)gsrc)
               : "memory");
}
__device__ inline void wait_async_le16() {
  asm volatile("s_wait_asynccnt 16" ::: "memory");
}
__device__ inline void wait_async_0() {
  asm volatile("s_wait_asynccnt 0" ::: "memory");
}

// ---- fp32 -> bf16 conversion ----
__global__ void cvt_f32_bf16(const float* __restrict__ src, __bf16* __restrict__ dst, int n) {
  int i = blockIdx.x * blockDim.x + threadIdx.x;
  if (i < n) dst[i] = (__bf16)src[i];
}

// ---- GEMM: out = A[NT,DM] @ W[DM,DM]^T + bias; 64x32 tile per wave (4x2 accs) ----
// mode 0: bf16 -> [B,H,S,DK] (Q,K)   mode 1: bf16 -> [B,H,DK,S] (V^T)
// mode 2: fp32 -> [NT,DM] (final output)
__global__ void gemm_bf16_kernel(const __bf16* __restrict__ A,
                                 const __bf16* __restrict__ W,
                                 const float*  __restrict__ bias,
                                 void* __restrict__ out, int mode) {
  const int lane  = threadIdx.x & 31;
  const int wave  = threadIdx.x >> 5;
  const int laneN = lane & 15;          // == laneM for A fragments
  const int half  = lane >> 4;
  // 4 waves per block arranged 2x2; each wave owns a 64x32 output tile
  const int m0 = (blockIdx.x * 2 + (wave >> 1)) * 64;
  const int n0 = (blockIdx.y * 2 + (wave & 1)) * 32;

  const __bf16* arow[4];
  const __bf16* brow[2];
#pragma unroll
  for (int r = 0; r < 4; r++)
    arow[r] = A + (size_t)(m0 + r * 16 + laneN) * DM + half * 16;
#pragma unroll
  for (int c = 0; c < 2; c++)
    brow[c] = W + (size_t)(n0 + c * 16 + laneN) * DM + half * 16;

  v8f acc[4][2] = {};
  for (int k0 = 0; k0 < DM; k0 += 32) {
    v16bf af[4], bf[2];
#pragma unroll
    for (int r = 0; r < 4; r++) af[r] = ld_fragA(arow[r] + k0);
#pragma unroll
    for (int c = 0; c < 2; c++) bf[c] = ld_fragB(brow[c] + k0);
#pragma unroll
    for (int r = 0; r < 4; r++)
#pragma unroll
      for (int c = 0; c < 2; c++)
        acc[r][c] = WMMA_BF16(af[r], bf[c], acc[r][c]);
  }

#pragma unroll
  for (int c = 0; c < 2; c++) {
    const int col = n0 + c * 16 + laneN;
    const float bcol = bias[col];
    if (mode == 2) {
      float* o = (float*)out;
#pragma unroll
      for (int r = 0; r < 4; r++)
#pragma unroll
        for (int j = 0; j < 8; j++) {
          int row = m0 + r * 16 + j + (half ? 8 : 0);
          o[(size_t)row * DM + col] = acc[r][c][j] + bcol;
        }
    } else {
      __bf16* o = (__bf16*)out;
      const int h = col >> 6, d = col & 63;
#pragma unroll
      for (int r = 0; r < 4; r++)
#pragma unroll
        for (int j = 0; j < 8; j++) {
          int row = m0 + r * 16 + j + (half ? 8 : 0);
          int b   = row >> 11;
          int s   = row & (S_ - 1);
          size_t idx = (mode == 0)
            ? ((size_t)((b * H_ + h) * S_ + s) * DK + d)     // [B,H,S,DK]
            : ((size_t)((b * H_ + h) * DK + d) * S_ + s);    // [B,H,DK,S]
          o[idx] = (__bf16)(acc[r][c][j] + bcol);
        }
    }
  }
}

// ---- flash attention: one wave per (b*H+h, 16-query tile) ----
// Q,K: [B,H,S,DK] bf16; Vt: [B,H,DK,S] bf16; O: [B,S,DM] bf16 (input to Wo GEMM)
// K/V 32-key tiles are double-buffered in LDS via async global->LDS copies.
__global__ void attn_kernel(const __bf16* __restrict__ Q,
                            const __bf16* __restrict__ K,
                            const __bf16* __restrict__ Vt,
                            __bf16* __restrict__ O) {
  __shared__ alignas(16) __bf16 ksm[2][32 * DK];   // 2 x 4KB: K tile, row-major [kk][d]
  __shared__ alignas(16) __bf16 vsm[2][DK * 32];   // 2 x 4KB: V tile, [d][kk]
  __shared__ alignas(16) __bf16 psm[16 * 32];      // P transpose buffer (1KB)

  const int lane  = threadIdx.x & 31;
  const int laneN = lane & 15;
  const int half  = lane >> 4;
  const int bh = blockIdx.x;          // b*H + h
  const int qt = blockIdx.y;
  const int q0 = qt * 16;

  const __bf16* qp = Q  + (size_t)bh * S_ * DK;
  const __bf16* kp = K  + (size_t)bh * S_ * DK;
  const __bf16* vp = Vt + (size_t)bh * DK * S_;

  // Q fragments (held for whole loop): d 0..31 and 32..63
  const __bf16* qrow = qp + (size_t)(q0 + laneN) * DK + half * 16;
  const v16bf aq0 = ld_fragA(qrow);
  const v16bf aq1 = ld_fragA(qrow + 32);

  // stage tile t (32 keys at t*32) into buffers ksm[buf]/vsm[buf]: 16 asyncs/lane
  auto stage = [&](int t, int buf) {
    const int k032 = t * 32;
    // K tile: 32 rows x 64 bf16 = 4KB, fully contiguous in memory
    const char* gk = (const char*)(kp + (size_t)k032 * DK);
    unsigned lk = (unsigned)(uintptr_t)&ksm[buf][0];
#pragma unroll
    for (int i = 0; i < 8; i++) {
      int off = (lane + i * 32) * 16;          // 0..4080 bytes
      async_copy_b128(lk + off, gk + off);
    }
    // V tile: 64 rows (d) x 32 bf16 (64B) -> vsm[d*32 + kk]
    unsigned lv = (unsigned)(uintptr_t)&vsm[buf][0];
#pragma unroll
    for (int i = 0; i < 8; i++) {
      int c  = lane + i * 32;                  // 0..255 16B chunks
      int d  = c >> 2;
      int ob = (c & 3) * 16;                   // byte offset within 64B row
      async_copy_b128(lv + d * 64 + ob,
                      (const char*)(vp + (size_t)d * S_ + k032) + ob);
    }
  };

  float m_run[8], l_run[8];
  v8f o0 = {}, o1 = {}, o2 = {}, o3 = {};
#pragma unroll
  for (int j = 0; j < 8; j++) { m_run[j] = -1e30f; l_run[j] = 0.0f; }

  const int ntile = (qt + 2) >> 1;    // ceil((qt+1)*16 / 32) causal 32-key tiles
  stage(0, 0);
  for (int t = 0; t < ntile; t++) {
    const int buf = t & 1;
    if (t + 1 < ntile) { stage(t + 1, buf ^ 1); wait_async_le16(); }
    else               { wait_async_0(); }

    const int k032 = t * 32;
    // scores for two 16-key subtiles from LDS K tile
    v8f sc0 = {}, sc1 = {};
    {
      const __bf16* kr = &ksm[buf][(size_t)laneN * DK + half * 16];
      sc0 = WMMA_BF16(aq0, ld_fragB(kr),      sc0);
      sc0 = WMMA_BF16(aq1, ld_fragB(kr + 32), sc0);
    }
    {
      const __bf16* kr = &ksm[buf][(size_t)(16 + laneN) * DK + half * 16];
      sc1 = WMMA_BF16(aq0, ld_fragB(kr),      sc1);
      sc1 = WMMA_BF16(aq1, ld_fragB(kr + 32), sc1);
    }

    // scale (1/sqrt(DK) = 0.125) + causal mask
    const int kg0 = k032 + laneN, kg1 = kg0 + 16;
#pragma unroll
    for (int j = 0; j < 8; j++) {
      const int qg = q0 + j + (half ? 8 : 0);
      sc0[j] = (kg0 > qg) ? -1e9f : sc0[j] * 0.125f;
      sc1[j] = (kg1 > qg) ? -1e9f : sc1[j] * 0.125f;
    }

    // online softmax: per-row reductions across each 16-lane half
#pragma unroll
    for (int j = 0; j < 8; j++) {
      float tm   = rmax16(fmaxf(sc0[j], sc1[j]));
      float mn   = fmaxf(m_run[j], tm);
      float p0   = __expf(sc0[j] - mn);
      float p1   = __expf(sc1[j] - mn);
      float rs   = rsum16(p0 + p1);
      float corr = __expf(m_run[j] - mn);
      l_run[j] = l_run[j] * corr + rs;
      m_run[j] = mn;
      o0[j] *= corr; o1[j] *= corr; o2[j] *= corr; o3[j] *= corr;
      const int row = j + (half ? 8 : 0);
      psm[row * 32 + laneN]      = (__bf16)p0;   // C-layout -> LDS
      psm[row * 32 + 16 + laneN] = (__bf16)p1;
    }
    __syncthreads();

    // P as A fragment (16x32) from LDS; V as B fragments from LDS V tile
    v16bf ap = ld_fragA(&psm[laneN * 32 + half * 16]);
    const __bf16* vr = &vsm[buf][(size_t)laneN * 32 + half * 16];
    o0 = WMMA_BF16(ap, ld_fragB(vr),            o0);
    o1 = WMMA_BF16(ap, ld_fragB(vr + 16 * 32),  o1);
    o2 = WMMA_BF16(ap, ld_fragB(vr + 32 * 32),  o2);
    o3 = WMMA_BF16(ap, ld_fragB(vr + 48 * 32),  o3);
    __syncthreads();
  }

  // epilogue: normalize by row sum, store O[b][s][h*64 + d] bf16
  const int bb = bh >> 4, hh = bh & 15;
#pragma unroll
  for (int j = 0; j < 8; j++) {
    const float inv = 1.0f / l_run[j];
    const int row = q0 + j + (half ? 8 : 0);
    const size_t base = ((size_t)bb * S_ + row) * DM + hh * DK + laneN;
    O[base +  0] = (__bf16)(o0[j] * inv);
    O[base + 16] = (__bf16)(o1[j] * inv);
    O[base + 32] = (__bf16)(o2[j] * inv);
    O[base + 48] = (__bf16)(o3[j] * inv);
  }
}

extern "C" void kernel_launch(void* const* d_in, const int* in_sizes, int n_in,
                              void* d_out, int out_size, void* d_ws, size_t ws_size,
                              hipStream_t stream) {
  const float* x  = (const float*)d_in[0];
  // d_in[1] = causal mask (structure known; not read)
  const float* Wq = (const float*)d_in[2];
  const float* bq = (const float*)d_in[3];
  const float* Wk = (const float*)d_in[4];
  const float* bk = (const float*)d_in[5];
  const float* Wv = (const float*)d_in[6];
  const float* bv = (const float*)d_in[7];
  const float* Wo = (const float*)d_in[8];
  const float* bo = (const float*)d_in[9];

  char* ws = (char*)d_ws;
  __bf16* xb  = (__bf16*)ws;                       //  8 MB: x bf16 [NT,DM]
  __bf16* wqb = (__bf16*)(ws + (8u  << 20));       //  2 MB each
  __bf16* wkb = wqb + (size_t)DM * DM;
  __bf16* wvb = wkb + (size_t)DM * DM;
  __bf16* wob = wvb + (size_t)DM * DM;
  __bf16* qb  = (__bf16*)(ws + (16u << 20));       //  8 MB: Q  [B,H,S,DK]
  __bf16* kb  = (__bf16*)(ws + (24u << 20));       //  8 MB: K  [B,H,S,DK]
  __bf16* vtb = (__bf16*)(ws + (32u << 20));       //  8 MB: V^T [B,H,DK,S]
  __bf16* ob  = (__bf16*)(ws + (40u << 20));       //  8 MB: attn out [B,S,DM]

  const int nX = NT * DM;        // 4,194,304
  const int nW = DM * DM;        // 1,048,576
  cvt_f32_bf16<<<nX / 256, 256, 0, stream>>>(x,  xb,  nX);
  cvt_f32_bf16<<<nW / 256, 256, 0, stream>>>(Wq, wqb, nW);
  cvt_f32_bf16<<<nW / 256, 256, 0, stream>>>(Wk, wkb, nW);
  cvt_f32_bf16<<<nW / 256, 256, 0, stream>>>(Wv, wvb, nW);
  cvt_f32_bf16<<<nW / 256, 256, 0, stream>>>(Wo, wob, nW);

  // 4 waves/block, each wave a 64x32 tile -> block covers 128x64
  dim3 ggrid(NT / 128, DM / 64);    // 32 x 16
  dim3 gblk(128);
  gemm_bf16_kernel<<<ggrid, gblk, 0, stream>>>(xb, wqb, bq, qb,  0);
  gemm_bf16_kernel<<<ggrid, gblk, 0, stream>>>(xb, wkb, bk, kb,  0);
  gemm_bf16_kernel<<<ggrid, gblk, 0, stream>>>(xb, wvb, bv, vtb, 1);

  attn_kernel<<<dim3(B_ * H_, S_ / 16), 32, 0, stream>>>(qb, kb, vtb, ob);

  gemm_bf16_kernel<<<ggrid, gblk, 0, stream>>>(ob, wob, bo, d_out, 2);
}
